// Interactor_32323923870246
// MI455X (gfx1250) — matrix operand order repaired and checked by reading, not compiled, not run on hardware
//
#include <hip/hip_runtime.h>

// ---------------------------------------------------------------------------
// Interactor graph network on MI455X (gfx1250, wave32, WMMA + TDM).
// * All 128-feature matmuls on v_wmma_f32_16x16x32_bf16 (fp32 accumulate).
// * Weights pre-packed once into a bf16 fragment arena (exact LDS image of the
//   wave32 B-fragment layout), then staged into LDS per block with
//   tensor_load_to_lds (TDM) + s_wait_tensorcnt.
// * GroupNorm / ReLU / residual fused into GEMM epilogues via wave shuffles.
// * lane_pooling edge path fully fused: pose MLP -> K=256 GEMM -> GN+ReLU ->
//   LDS transpose -> K=128 GEMM -> atomic f32 scatter-add.
// ---------------------------------------------------------------------------

#define NG 50000
#define NR 12000
#define CF 128
#define EPS_N 50000
#define ELR_N 5000
#define E1_N 150000
#define E2_N 150000

typedef __bf16 bf16;
typedef __attribute__((ext_vector_type(16))) __bf16 v16bf;
typedef __attribute__((ext_vector_type(8)))  float  v8f;
typedef unsigned int u32x4 __attribute__((ext_vector_type(4)));
typedef int          i32x8 __attribute__((ext_vector_type(8)));
typedef int          i32x4 __attribute__((ext_vector_type(4)));

// ---------------------------------------------------------------------------
// TDM: copy nbytes (4B-multiple, <= 65535 dwords) from global to LDS as a 1-D
// tile.  D# per CDNA5 ISA: group0 = {count|flags,
// lds_addr, global_addr, type=2}; group1 = dims/strides, data_size=4B.
// 6-arg builtin on this toolchain (groups 2/3 + trailing i32x8 all zero for
// a <=2D tensor).  Issued by one wave; completion via s_wait_tensorcnt 0.
// ---------------------------------------------------------------------------
__device__ __forceinline__ void tdm_copy_to_lds(unsigned lds_addr,
                                                const void* gsrc,
                                                unsigned nbytes) {
  unsigned n = nbytes >> 2;  // dword elements
  unsigned long long ga = (unsigned long long)(uintptr_t)gsrc;
  u32x4 g0;
  g0[0] = 1u;                                   // count=1 (valid), user mode
  g0[1] = lds_addr;                             // LDS byte address
  g0[2] = (unsigned)ga;                         // global_addr[31:0]
  g0[3] = (unsigned)((ga >> 32) & 0x1FFFFFFu)   // global_addr[56:32]
          | 0x80000000u;                        // type=2 ("image")
  i32x8 g1;
  g1[0] = (int)(2u << 16);                      // data_size = 4 bytes
  g1[1] = (int)((n & 0xFFFFu) << 16);           // tensor_dim0[15:0]
  g1[2] = (int)(((n >> 16) & 0xFFFFu)           // tensor_dim0[31:16]
          | (1u << 16));                        // tensor_dim1 = 1
  g1[3] = (int)((n & 0xFFFFu) << 16);           // tile_dim0 = n
  g1[4] = 1;                                    // tile_dim1 = 1
  g1[5] = (int)n;                               // tensor_dim0_stride[31:0]
  g1[6] = 0;
  g1[7] = 0;
  i32x4 z4 = {};
  i32x8 z8 = {};
  __builtin_amdgcn_tensor_load_to_lds(g0, g1, z4, z4, z8, 0);
}

// Generic-LDS pointer -> LDS byte offset (LDS offset is the low 32 bits of the address).
__device__ __forceinline__ unsigned lds_off(const void* p) {
  return (unsigned)(uintptr_t)p;
}

// ---------------------------------------------------------------------------
// Pre-pack W (row-major [128][K], out = X @ W^T) into the wave32 WMMA
// B-fragment LDS image: linear idx -> (s = kstep, t = col tile, lane, i):
//   dst[((s*8 + t)*32 + lane)*16 + i] =
//     bf16( W[(16t+(lane&15))*K + 32s + 16(i>>3) + 8(lane>>4) + (i&7)] )
// so B[k][n] = W[n][k] with lane&15 = n within 16-col tile, k in table order.
// ---------------------------------------------------------------------------
__global__ void prepack_w(const float* __restrict__ W, bf16* __restrict__ dst,
                          int K) {
  int idx = blockIdx.x * 256 + threadIdx.x;
  if (idx >= K * CF) return;
  int s  = idx >> 12;
  int r  = idx & 4095;
  int t  = r >> 9;
  int ln = (r >> 4) & 31;
  int i  = r & 15;
  dst[idx] = (bf16)W[(16 * t + (ln & 15)) * K + 32 * s + ((i >> 3) << 4) +
                     ((ln >> 4) << 3) + (i & 7)];
}

// Per-row GroupNorm(1 group over 128 ch) on the wave's 16x128 accumulator.
// Row m = g + 8*(lane>=16); col = 16*t + (lane&15).  xor-shuffles (masks 1..8)
// reduce within each 16-lane half.
__device__ __forceinline__ void gn_apply(v8f* acc, const float* __restrict__ sb,
                                         int lane, bool do_relu) {
  float mean[8], rs[8];
#pragma unroll
  for (int g = 0; g < 8; ++g) {
    float s1 = 0.f, s2 = 0.f;
#pragma unroll
    for (int t = 0; t < 8; ++t) { float x = acc[t][g]; s1 += x; s2 += x * x; }
#pragma unroll
    for (int off = 8; off >= 1; off >>= 1) {
      s1 += __shfl_xor(s1, off, 32);
      s2 += __shfl_xor(s2, off, 32);
    }
    float mn = s1 * 0.0078125f;
    float vr = s2 * 0.0078125f - mn * mn;
    mean[g] = mn;
    rs[g]   = rsqrtf(vr + 1e-5f);
  }
#pragma unroll
  for (int t = 0; t < 8; ++t) {
    int col  = 16 * t + (lane & 15);
    float sc = sb[col];
    float bi = sb[CF + col];
#pragma unroll
    for (int g = 0; g < 8; ++g) {
      float y = (acc[t][g] - mean[g]) * rs[g] * sc + bi;
      acc[t][g] = do_relu ? fmaxf(y, 0.f) : y;
    }
  }
}

// ---------------------------------------------------------------------------
// Dense GEMM: Y[N,128] = X[N,128] @ W^T.  Packed weights TDM-staged into LDS.
// EPI 0: raw.  EPI 1: relu(gn(.)).  EPI 2: relu(gn(.)+ident).
// 256 threads = 8 waves, 128 rows/block, 16 rows/wave.
// ---------------------------------------------------------------------------
template <int EPI>
__global__ void gemm128(const float* __restrict__ X, const bf16* __restrict__ Wp,
                        float* Y, const float* __restrict__ sb,
                        const float* ident, int N) {
  __shared__ __align__(32) bf16 Bps[4][8][32][16];   // 32 KB weight image
  int tid  = threadIdx.x;
  int lane = tid & 31;
  int wave = tid >> 5;
  if (wave == 0) {
    tdm_copy_to_lds(lds_off(&Bps[0][0][0][0]), Wp, 4 * 8 * 32 * 16 * 2);
    __builtin_amdgcn_s_wait_tensorcnt(0);
  }
  __syncthreads();

  int row0  = blockIdx.x * 128 + wave * 16;
  int m     = lane & 15;
  int rowm  = row0 + m;
  bool valid = rowm < N;
  const float* arow = X + (size_t)(valid ? rowm : 0) * CF;
  if (valid) {
    __builtin_prefetch(arow, 0, 0);
    __builtin_prefetch(arow + 64, 0, 0);
  }
  int kb = (lane >> 4) << 3;                   // A-layout half-select

  v8f acc[8] = {};
#pragma unroll
  for (int s = 0; s < 4; ++s) {
    v16bf a;
    int k0 = 32 * s + kb;
#pragma unroll
    for (int i = 0; i < 8; ++i) {
      a[i]     = valid ? (bf16)arow[k0 + i]      : (bf16)0.f;
      a[i + 8] = valid ? (bf16)arow[k0 + 16 + i] : (bf16)0.f;
    }
#pragma unroll
    for (int t = 0; t < 8; ++t) {
      v16bf b = *(const v16bf*)&Bps[s][t][lane][0];
      acc[t] = __builtin_amdgcn_wmma_f32_16x16x32_bf16(
          false, a, false, b, (short)0, acc[t], false, false);
    }
  }

  if (EPI >= 1) gn_apply(acc, sb, lane, EPI == 1);

#pragma unroll
  for (int t = 0; t < 8; ++t) {
    int col = 16 * t + (lane & 15);
#pragma unroll
    for (int g = 0; g < 8; ++g) {
      int r = row0 + g + ((lane >> 4) << 3);
      if (r < N) {
        float y = acc[t][g];
        if (EPI == 2) {
          float idv = ident ? ident[(size_t)r * CF + col] : 0.f;
          y = fmaxf(y + idv, 0.f);
        }
        Y[(size_t)r * CF + col] = y;
      }
    }
  }
}

// ---------------------------------------------------------------------------
// Fused lane-pooling edge kernel.  One wave = 16 edges.
//   dist = relu((ctx_pose[hi]-tgt_pose[wi]) @ Wr^T + br)    (in-regs)
//   h1   = relu(gn(concat(ctx_feat[hi], dist) @ W1^T, gn1)) (K=256 WMMA)
//   y    = h1 @ W2^T                                        (K=128 WMMA)
//   T[wi] += y  (atomic f32 scatter)
// Packed W1/W2 staged in 16 KB TDM chunks (2 K-steps each).
// ---------------------------------------------------------------------------
__global__ void lane_pool_edges(const float* __restrict__ ctx_feat,
                                const float* __restrict__ ctx_pose,
                                const float* __restrict__ tgt_pose,
                                const int* __restrict__ hi,
                                const int* __restrict__ wi,
                                const float* __restrict__ Wr,
                                const float* __restrict__ br,
                                const bf16* __restrict__ W1p,
                                const float* __restrict__ gn1,
                                const bf16* __restrict__ W2p,
                                float* T, int E) {
  __shared__ __align__(32) bf16 Bp[2][8][32][16];    // 16 KB staging chunk
  __shared__ __align__(16) bf16 Hst[8][16][CF];      // 32 KB per-wave H1 tiles
  __shared__ float4 Wrl[CF];
  __shared__ float  brl[CF];
  __shared__ int    wib[8][16];

  int tid  = threadIdx.x;
  int lane = tid & 31;
  int wave = tid >> 5;
  if (tid < CF) { Wrl[tid] = ((const float4*)Wr)[tid]; brl[tid] = br[tid]; }

  int eb = blockIdx.x * 128 + wave * 16;
  int m  = lane & 15;
  int e  = eb + m;
  bool ev = e < E;
  int h = ev ? hi[e] : 0;
  int w = ev ? wi[e] : 0;
  if (lane < 16) wib[wave][lane] = ev ? w : -1;

  const float* arow = ctx_feat + (size_t)h * CF;
  if (ev) {                                   // warm the gathered row (512 B)
    __builtin_prefetch(arow, 0, 0);
    __builtin_prefetch(arow + 32, 0, 0);
    __builtin_prefetch(arow + 64, 0, 0);
    __builtin_prefetch(arow + 96, 0, 0);
  }

  float4 cp = ev ? ((const float4*)ctx_pose)[h] : make_float4(0.f, 0.f, 0.f, 0.f);
  float4 tp = ev ? ((const float4*)tgt_pose)[w] : make_float4(0.f, 0.f, 0.f, 0.f);
  float pd0 = cp.x - tp.x, pd1 = cp.y - tp.y, pd2 = cp.z - tp.z, pd3 = cp.w - tp.w;

  int kb = (lane >> 4) << 3;
  unsigned bp_lds = lds_off(&Bp[0][0][0][0]);

  // ---- GEMM1: K = 256 (first 128 = gathered feat, last 128 = dist MLP) ----
  v8f acc[8] = {};
  for (int sp = 0; sp < 4; ++sp) {
    __syncthreads();                           // prior chunk fully consumed
    if (wave == 0) {
      tdm_copy_to_lds(bp_lds, W1p + sp * 8192, 16384);
      __builtin_amdgcn_s_wait_tensorcnt(0);
    }
    __syncthreads();
#pragma unroll
    for (int s2 = 0; s2 < 2; ++s2) {
      int s = 2 * sp + s2;
      v16bf a;
      if (sp < 2) {                            // gathered ctx_feat half
        int k0 = 32 * s + kb;
#pragma unroll
        for (int i = 0; i < 8; ++i) {
          a[i]     = ev ? (bf16)arow[k0 + i]      : (bf16)0.f;
          a[i + 8] = ev ? (bf16)arow[k0 + 16 + i] : (bf16)0.f;
        }
      } else {                                 // dist-MLP half, in-regs
        int c0 = 32 * (s - 4);
#pragma unroll
        for (int i = 0; i < 16; ++i) {
          int c = c0 + ((i >> 3) << 4) + kb + (i & 7);
          float4 wr = Wrl[c];
          float dv = fmaxf(pd0 * wr.x + pd1 * wr.y + pd2 * wr.z + pd3 * wr.w
                           + brl[c], 0.f);
          a[i] = ev ? (bf16)dv : (bf16)0.f;
        }
      }
#pragma unroll
      for (int t = 0; t < 8; ++t) {
        v16bf b = *(const v16bf*)&Bp[s2][t][lane][0];
        acc[t] = __builtin_amdgcn_wmma_f32_16x16x32_bf16(
            false, a, false, b, (short)0, acc[t], false, false);
      }
    }
  }

  gn_apply(acc, gn1, lane, true);              // relu(gn(ctx1))

  // ---- transpose H1 (C-layout -> row-major bf16) via wave-private LDS -----
#pragma unroll
  for (int t = 0; t < 8; ++t) {
    int col = 16 * t + (lane & 15);
#pragma unroll
    for (int g = 0; g < 8; ++g) {
      int mr = g + ((lane >> 4) << 3);
      Hst[wave][mr][col] = (bf16)acc[t][g];
    }
  }

  // ---- GEMM2: K = 128, A from LDS -----------------------------------------
  v8f acc2[8] = {};
  const bf16* hrow = &Hst[wave][m][0];
  for (int sp = 0; sp < 2; ++sp) {
    __syncthreads();                           // also publishes Hst on sp==0
    if (wave == 0) {
      tdm_copy_to_lds(bp_lds, W2p + sp * 8192, 16384);
      __builtin_amdgcn_s_wait_tensorcnt(0);
    }
    __syncthreads();
#pragma unroll
    for (int s2 = 0; s2 < 2; ++s2) {
      int s = 2 * sp + s2;
      int k0 = 32 * s + kb;
      v16bf a;
#pragma unroll
      for (int i = 0; i < 8; ++i) {
        a[i]     = hrow[k0 + i];
        a[i + 8] = hrow[k0 + 16 + i];
      }
#pragma unroll
      for (int t = 0; t < 8; ++t) {
        v16bf b = *(const v16bf*)&Bp[s2][t][lane][0];
        acc2[t] = __builtin_amdgcn_wmma_f32_16x16x32_bf16(
            false, a, false, b, (short)0, acc2[t], false, false);
      }
    }
  }

  // ---- atomic scatter-add onto target nodes -------------------------------
#pragma unroll
  for (int t = 0; t < 8; ++t) {
    int col = 16 * t + (lane & 15);
#pragma unroll
    for (int g = 0; g < 8; ++g) {
      int mr = g + ((lane >> 4) << 3);
      int wd = wib[wave][mr];
      if (wd >= 0) atomicAdd(&T[(size_t)wd * CF + col], acc2[t][g]);
    }
  }
}

// Y[r] = relu(gn(X[r], sb)); one wave per row, 4 cols/lane.
__global__ void rowgn_relu(const float* __restrict__ X, const float* __restrict__ sb,
                           float* __restrict__ Y, int N) {
  int r    = blockIdx.x * 8 + (threadIdx.x >> 5);
  int lane = threadIdx.x & 31;
  if (r >= N) return;
  float4 x = ((const float4*)(X + (size_t)r * CF))[lane];
  float s1 = x.x + x.y + x.z + x.w;
  float s2 = x.x * x.x + x.y * x.y + x.z * x.z + x.w * x.w;
#pragma unroll
  for (int off = 16; off >= 1; off >>= 1) {
    s1 += __shfl_xor(s1, off, 32);
    s2 += __shfl_xor(s2, off, 32);
  }
  float mn = s1 * 0.0078125f;
  float rs = rsqrtf(s2 * 0.0078125f - mn * mn + 1e-5f);
  float4 sc = ((const float4*)sb)[lane];
  float4 bi = ((const float4*)(sb + CF))[lane];
  float4 y;
  y.x = fmaxf((x.x - mn) * rs * sc.x + bi.x, 0.f);
  y.y = fmaxf((x.y - mn) * rs * sc.y + bi.y, 0.f);
  y.z = fmaxf((x.z - mn) * rs * sc.z + bi.z, 0.f);
  y.w = fmaxf((x.w - mn) * rs * sc.w + bi.w, 0.f);
  ((float4*)(Y + (size_t)r * CF))[lane] = y;
}

// T[u[e]] += XW[v[e]]  (row scatter-add; 32 threads/edge, 4 floats/thread)
__global__ void scatter_add(const float* __restrict__ XW, const int* __restrict__ u,
                            const int* __restrict__ v, float* T, int E) {
  int idx = blockIdx.x * blockDim.x + threadIdx.x;
  int e = idx >> 5;
  if (e >= E) return;
  int c = (idx & 31) * 4;
  int uu = u[e], vv = v[e];
  float4 s = *(const float4*)(XW + (size_t)vv * CF + c);
  float* d = T + (size_t)uu * CF + c;
  atomicAdd(d + 0, s.x);
  atomicAdd(d + 1, s.y);
  atomicAdd(d + 2, s.z);
  atomicAdd(d + 3, s.w);
}

// ---------------------------------------------------------------------------
extern "C" void kernel_launch(void* const* d_in, const int* in_sizes, int n_in,
                              void* d_out, int out_size, void* d_ws, size_t ws_size,
                              hipStream_t stream) {
  const float* roi_feat     = (const float*)d_in[0];
  const float* graph_pose   = (const float*)d_in[1];
  const float* roi_pose     = (const float*)d_in[2];
  const float* lp_input_w   = (const float*)d_in[3];
  const float* lp_relpose_w = (const float*)d_in[4];
  const float* lp_relpose_b = (const float*)d_in[5];
  const float* lp_ctx1_w    = (const float*)d_in[6];
  const float* lp_ctx1_gn   = (const float*)d_in[7];
  const float* lp_ctx2_w    = (const float*)d_in[8];
  const float* lp_mlp1_w    = (const float*)d_in[9];
  const float* lp_mlp1_gn   = (const float*)d_in[10];
  const float* lp_mlp2_w    = (const float*)d_in[11];
  const float* lp_mlp2_gn   = (const float*)d_in[12];
  const float* lp_norm_gn   = (const float*)d_in[13];
  const float* fuse_ctr_w   = (const float*)d_in[14];
  const float* fuse_ps_w    = (const float*)d_in[15];
  const float* fuse_left_w  = (const float*)d_in[16];
  const float* fuse_right_w = (const float*)d_in[17];
  const float* fuse_norm_gn = (const float*)d_in[18];
  const float* fuse_ctr2_w  = (const float*)d_in[19];
  const float* fuse_ctr2_gn = (const float*)d_in[20];
  const int*   e1_hi = (const int*)d_in[21];
  const int*   e1_wi = (const int*)d_in[22];
  const int*   e2_hi = (const int*)d_in[23];
  const int*   e2_wi = (const int*)d_in[24];
  const int*   ps_u  = (const int*)d_in[25];
  const int*   ps_v  = (const int*)d_in[26];
  const int*   lr_u  = (const int*)d_in[27];
  const int*   lr_v  = (const int*)d_in[28];

  const size_t NGC = (size_t)NG * CF;
  const size_t CC  = (size_t)CF * CF;
  float* GF   = (float*)d_ws;          // graph features
  float* TEMP = GF + NGC;              // scatter accumulator
  float* XW   = TEMP + NGC;            // transient dense-GEMM outputs
  float* OUT  = (float*)d_out;
  bf16*  AR   = (bf16*)(XW + NGC);     // packed-weight arena (~2.5 MB)

  // ---- pre-pack all weights into B-fragment arenas ------------------------
  size_t aoff = 0;
  auto PK = [&](const float* W, int K) -> const bf16* {
    bf16* dptr = AR + aoff;
    int n = K * CF;
    prepack_w<<<(n + 255) / 256, 256, 0, stream>>>(W, dptr, K);
    aoff += (size_t)n;
    return dptr;
  };
  const bf16* P_ctx1_0 = PK(lp_ctx1_w, 256);
  const bf16* P_ctx2_0 = PK(lp_ctx2_w, 128);
  const bf16* P_mlp1_0 = PK(lp_mlp1_w, 128);
  const bf16* P_mlp2_0 = PK(lp_mlp2_w, 128);
  const bf16* P_ctr[4];
  const bf16* P_ps[48];
  const bf16* P_left[4];
  const bf16* P_right[4];
  const bf16* P_ctr2[4];
  for (int i = 0; i < 4; ++i) P_ctr[i]   = PK(fuse_ctr_w + (size_t)i * CC, 128);
  for (int i = 0; i < 4; ++i)
    for (int k = 0; k < 12; ++k)
      P_ps[i * 12 + k] = PK(fuse_ps_w + ((size_t)i * 12 + k) * CC, 128);
  for (int i = 0; i < 4; ++i) P_left[i]  = PK(fuse_left_w + (size_t)i * CC, 128);
  for (int i = 0; i < 4; ++i) P_right[i] = PK(fuse_right_w + (size_t)i * CC, 128);
  for (int i = 0; i < 4; ++i) P_ctr2[i]  = PK(fuse_ctr2_w + (size_t)i * CC, 128);
  const bf16* P_inp1   = PK(lp_input_w + CC, 128);
  const bf16* P_ctx1_1 = PK(lp_ctx1_w + (size_t)CF * 2 * CF, 256);
  const bf16* P_ctx2_1 = PK(lp_ctx2_w + CC, 128);
  const bf16* P_mlp1_1 = PK(lp_mlp1_w + CC, 128);
  const bf16* P_mlp2_1 = PK(lp_mlp2_w + CC, 128);

  auto gb = [](int n) { return (n + 127) / 128; };

  // ---------------- lane_pooling #1 (roi -> graph; tgt_feat == 0) ----------
  (void)hipMemsetAsync(TEMP, 0, NGC * sizeof(float), stream);
  lane_pool_edges<<<gb(E1_N), 256, 0, stream>>>(
      roi_feat, roi_pose, graph_pose, e1_hi, e1_wi,
      lp_relpose_w, lp_relpose_b, P_ctx1_0, lp_ctx1_gn, P_ctx2_0, TEMP, E1_N);
  rowgn_relu<<<(NG + 7) / 8, 256, 0, stream>>>(TEMP, lp_norm_gn, XW, NG);
  gemm128<1><<<gb(NG), 256, 0, stream>>>(XW, P_mlp1_0, TEMP, lp_mlp1_gn, nullptr, NG);
  gemm128<2><<<gb(NG), 256, 0, stream>>>(TEMP, P_mlp2_0, GF, lp_mlp2_gn, nullptr, NG);

  // ---------------- global_graph: 4 fusion layers --------------------------
  for (int i = 0; i < 4; ++i) {
    gemm128<0><<<gb(NG), 256, 0, stream>>>(GF, P_ctr[i], TEMP, nullptr, nullptr, NG);
    for (int k = 0; k < 12; ++k) {
      gemm128<0><<<gb(NG), 256, 0, stream>>>(GF, P_ps[i * 12 + k], XW,
                                             nullptr, nullptr, NG);
      scatter_add<<<(EPS_N * 32 + 255) / 256, 256, 0, stream>>>(
          XW, ps_u + (size_t)k * EPS_N, ps_v + (size_t)k * EPS_N, TEMP, EPS_N);
    }
    gemm128<0><<<gb(NG), 256, 0, stream>>>(GF, P_left[i], XW, nullptr, nullptr, NG);
    scatter_add<<<(ELR_N * 32 + 255) / 256, 256, 0, stream>>>(XW, lr_u, lr_v,
                                                              TEMP, ELR_N);
    gemm128<0><<<gb(NG), 256, 0, stream>>>(GF, P_right[i], XW, nullptr, nullptr, NG);
    scatter_add<<<(ELR_N * 32 + 255) / 256, 256, 0, stream>>>(XW, lr_u + ELR_N,
                                                              lr_v + ELR_N,
                                                              TEMP, ELR_N);
    rowgn_relu<<<(NG + 7) / 8, 256, 0, stream>>>(
        TEMP, fuse_norm_gn + (size_t)i * 2 * CF, XW, NG);
    // GF = relu(gn(XW @ ctr2^T) + GF)
    gemm128<2><<<gb(NG), 256, 0, stream>>>(XW, P_ctr2[i], GF,
                                           fuse_ctr2_gn + (size_t)i * 2 * CF,
                                           GF, NG);
  }

  // ---------------- lane_pooling #2 (graph -> roi) -------------------------
  gemm128<0><<<gb(NR), 256, 0, stream>>>(roi_feat, P_inp1, TEMP,
                                         nullptr, nullptr, NR);
  lane_pool_edges<<<gb(E2_N), 256, 0, stream>>>(
      GF, graph_pose, roi_pose, e2_hi, e2_wi,
      lp_relpose_w + (size_t)CF * 4, lp_relpose_b + CF,
      P_ctx1_1, lp_ctx1_gn + 2 * CF, P_ctx2_1, TEMP, E2_N);
  rowgn_relu<<<(NR + 7) / 8, 256, 0, stream>>>(TEMP, lp_norm_gn + 2 * CF, XW, NR);
  gemm128<1><<<gb(NR), 256, 0, stream>>>(XW, P_mlp1_1, TEMP,
                                         lp_mlp1_gn + 2 * CF, nullptr, NR);
  gemm128<2><<<gb(NR), 256, 0, stream>>>(TEMP, P_mlp2_1, OUT,
                                         lp_mlp2_gn + 2 * CF, roi_feat, NR);
}
